// multiHeadSelfAttention_70909910057012
// MI455X (gfx1250) — matrix operand order repaired and checked by reading, not compiled, not run on hardware
//
#include <hip/hip_runtime.h>

// ---- problem constants ----
#define B_   8
#define S_   1024
#define H_   16
#define DH_  64
#define D_   768     // input feature dim
#define OD_  1024    // H_*DH_
#define N3_  3072    // 3*OD_
#define M_   8192    // B_*S_

typedef __bf16 bf16;
typedef __attribute__((ext_vector_type(16))) __bf16 v16bf;
typedef __attribute__((ext_vector_type(8)))  __bf16 v8bf;
typedef __attribute__((ext_vector_type(4)))  __bf16 v4bf;
typedef __attribute__((ext_vector_type(8)))  float  v8f;

union Frag16 { v16bf v; v8bf h[2]; };

__device__ __forceinline__ bf16 f2bf(float f) {
  unsigned u = __builtin_bit_cast(unsigned, f);
  u += 0x7fffu + ((u >> 16) & 1u);           // round-to-nearest-even
  unsigned short hs = (unsigned short)(u >> 16);
  return __builtin_bit_cast(bf16, hs);
}

// CDNA5 async global->LDS copy (16B per lane), tracked by ASYNCcnt.
// Low 32 bits of a flat pointer to __shared__ are the LDS byte address.
__device__ __forceinline__ void async_copy16(void* lds_ptr, const bf16* gptr) {
  unsigned loff = (unsigned)(size_t)lds_ptr;
  asm volatile("global_load_async_to_lds_b128 %0, %1, off"
               :: "v"(loff), "v"(gptr) : "memory");
}
__device__ __forceinline__ void wait_async0() {
  asm volatile("s_wait_asynccnt 0x0" ::: "memory");
}

// ---------------- kernel 1: x (f32) -> bf16 ----------------
__global__ __launch_bounds__(256) void k_conv_x(const float* __restrict__ x,
                                                bf16* __restrict__ xb) {
  int i = (blockIdx.x * 256 + threadIdx.x) * 4;
  float4 f = *reinterpret_cast<const float4*>(x + i);
  v4bf o;
  o[0] = f2bf(f.x); o[1] = f2bf(f.y); o[2] = f2bf(f.z); o[3] = f2bf(f.w);
  *reinterpret_cast<v4bf*>(xb + i) = o;
}

// -------- kernel 2: weights -> wt[N3_][D_] bf16 (transposed, q scaled) --------
__global__ __launch_bounds__(256) void k_conv_w(const float* __restrict__ qw,
                                                const float* __restrict__ kw,
                                                const float* __restrict__ vw,
                                                bf16* __restrict__ wt) {
  int idx  = blockIdx.x * 256 + threadIdx.x;        // [0, 3*D_*OD_)
  int widx = idx / (D_ * OD_);
  int e    = idx % (D_ * OD_);
  int k    = e % D_;
  int n    = e / D_;
  const float* w = (widx == 0) ? qw : ((widx == 1) ? kw : vw);
  float scale = (widx == 0) ? 0.125f : 1.0f;        // fold 1/sqrt(Dh) into Q
  wt[(size_t)(widx * OD_ + n) * D_ + k] = f2bf(w[(size_t)k * OD_ + n] * scale);
}

// ---------------- kernel 3: fused QKV GEMM (bf16 WMMA, async->LDS, 2x buffered) ----------------
__global__ __launch_bounds__(256) void k_qkv_gemm(const bf16* __restrict__ xb,
                                                  const bf16* __restrict__ wt,
                                                  bf16* __restrict__ Qb,
                                                  bf16* __restrict__ Kb,
                                                  bf16* __restrict__ Vt) {
  __shared__ __align__(16) bf16 As[2][128 * 40];
  __shared__ __align__(16) bf16 Bs[2][128 * 40];
  const int tid  = threadIdx.x;
  const int lane = tid & 31, w = tid >> 5;
  const int half = lane >> 4, lr = lane & 15;
  const int m0 = blockIdx.x * 128, n0 = blockIdx.y * 128;
  const int wm = (w & 3) * 32, wn = (w >> 2) * 64;

  const v8f vzero = {};
  v8f acc[2][4];
#pragma unroll
  for (int mi = 0; mi < 2; ++mi)
#pragma unroll
    for (int ni = 0; ni < 4; ++ni) acc[mi][ni] = vzero;

  // cooperative stage of one 128x32 A tile + 128x32 B tile via async copies
  auto stage = [&](int kk, int buf) {
#pragma unroll
    for (int pass = 0; pass < 2; ++pass) {
      int t = tid + pass * 256;
      int row = t >> 2, ch = (t & 3) * 8;
      async_copy16(&As[buf][row * 40 + ch],
                   &xb[(size_t)(m0 + row) * D_ + kk + ch]);
      async_copy16(&Bs[buf][row * 40 + ch],
                   &wt[(size_t)(n0 + row) * D_ + kk + ch]);
    }
  };

  auto compute = [&](int buf) {
    Frag16 a[2], b[4];
#pragma unroll
    for (int mi = 0; mi < 2; ++mi) {   // A 16x32: lane<16 K{0-7,16-23}, else K{8-15,24-31}
      int r = wm + mi * 16 + lr;
      int ka = half * 8;
      a[mi].h[0] = *reinterpret_cast<const v8bf*>(&As[buf][r * 40 + ka]);
      a[mi].h[1] = *reinterpret_cast<const v8bf*>(&As[buf][r * 40 + ka + 16]);
    }
#pragma unroll
    for (int ni = 0; ni < 4; ++ni) {   // B 32x16: lane<16 K{0-7,16-23}, else K{8-15,24-31}
      int r = wn + ni * 16 + lr;
      int kb = half * 8;
      b[ni].h[0] = *reinterpret_cast<const v8bf*>(&Bs[buf][r * 40 + kb]);
      b[ni].h[1] = *reinterpret_cast<const v8bf*>(&Bs[buf][r * 40 + kb + 16]);
    }
#pragma unroll
    for (int mi = 0; mi < 2; ++mi)
#pragma unroll
      for (int ni = 0; ni < 4; ++ni)
        acc[mi][ni] = __builtin_amdgcn_wmma_f32_16x16x32_bf16(
            false, a[mi].v, false, b[ni].v, (short)0, acc[mi][ni], false, false);
  };

  const int NK = D_ / 32;   // 24 (even)
  stage(0, 0);
  wait_async0();
  __syncthreads();

  // branchless steady state, literal buffer indices (no phi copies)
  for (int kt = 0; kt < NK - 2; kt += 2) {
    stage((kt + 1) * 32, 1);
    compute(0);
    wait_async0(); __syncthreads();
    stage((kt + 2) * 32, 0);
    compute(1);
    wait_async0(); __syncthreads();
  }
  stage((NK - 1) * 32, 1);
  compute(0);
  wait_async0(); __syncthreads();
  compute(1);

  // epilogue: Q,K row-major bf16; V transposed [b][h][dh][s]
#pragma unroll
  for (int mi = 0; mi < 2; ++mi) {
    int gm0 = m0 + wm + mi * 16 + half * 8;   // 8 consecutive rows per lane-half
#pragma unroll
    for (int ni = 0; ni < 4; ++ni) {
      int gn = n0 + wn + ni * 16 + lr;
      if (gn < 2 * OD_) {
        bf16* dst = (gn < OD_) ? Qb : Kb;
        int col = gn & (OD_ - 1);
#pragma unroll
        for (int j = 0; j < 8; ++j)
          dst[(size_t)(gm0 + j) * OD_ + col] = f2bf(acc[mi][ni][j]);
      } else {
        int od = gn - 2 * OD_;
        int bb = gm0 >> 10, s = gm0 & 1023;
        int hh = od >> 6, dh = od & 63;
        v8bf o;
#pragma unroll
        for (int j = 0; j < 8; ++j) o[j] = f2bf(acc[mi][ni][j]);
        *reinterpret_cast<v8bf*>(&Vt[((size_t)((bb * H_ + hh) * DH_ + dh)) * S_ + s]) = o;
      }
    }
  }
}

// ---------------- kernel 4: flash attention (bf16 WMMA, LDS-staged K/V) ----------------
__global__ __launch_bounds__(256) void k_attn(const bf16* __restrict__ Qb,
                                              const bf16* __restrict__ Kb,
                                              const bf16* __restrict__ Vt,
                                              float* __restrict__ out) {
  __shared__ __align__(16) bf16 Ks[2][32 * 72];    // 32 keys x 64 dh (+pad)
  __shared__ __align__(16) bf16 Vs[2][64 * 40];    // 64 dh x 32 keys (+pad)
  __shared__ __align__(16) bf16 pbuf[8][16 * 40];  // per-wave P transpose tile
  const int tid  = threadIdx.x;
  const int lane = tid & 31, w = tid >> 5;
  const int half = lane >> 4, lr = lane & 15;
  const int bh = blockIdx.x;               // b*H + h
  const int b = bh >> 4, h = bh & 15;
  const int q0 = blockIdx.y * 128 + w * 16;

  // cooperative async stage of one 32-key K/V chunk (shared by all 8 waves)
  auto stage = [&](int kb, int buf) {
    {  // K: 32 rows x 128B -> 8 lanes/row
      int r = tid >> 3, ch = (tid & 7) * 8;
      async_copy16(&Ks[buf][r * 72 + ch],
                   &Kb[(size_t)(b * S_ + kb + r) * OD_ + h * DH_ + ch]);
    }
    {  // V^T: 64 rows x 64B -> 4 lanes/row
      int r = tid >> 2, ch = (tid & 3) * 8;
      async_copy16(&Vs[buf][r * 40 + ch],
                   &Vt[((size_t)((b * H_ + h) * DH_ + r)) * S_ + kb + ch]);
    }
  };

  // Q fragments (scale already folded into q_w)
  Frag16 aq[2];
  {
    const size_t rowbase = ((size_t)(b * S_ + q0 + lr)) * OD_ + h * DH_;
#pragma unroll
    for (int f = 0; f < 2; ++f) {
      int k0 = f * 32 + half * 8;
      aq[f].h[0] = *reinterpret_cast<const v8bf*>(&Qb[rowbase + k0]);
      aq[f].h[1] = *reinterpret_cast<const v8bf*>(&Qb[rowbase + k0 + 16]);
    }
  }

  const v8f vzero = {};
  v8f o[4];
#pragma unroll
  for (int d = 0; d < 4; ++d) o[d] = vzero;
  float mrow[8], lrow[8];
#pragma unroll
  for (int j = 0; j < 8; ++j) { mrow[j] = -1e30f; lrow[j] = 0.0f; }

  bf16* pb = pbuf[w];

  // one full 32-key chunk: scores -> online softmax -> P transpose -> O += P*V
  auto chunk = [&](int buf) {
    v8f sc[2];
#pragma unroll
    for (int c = 0; c < 2; ++c) {
      sc[c] = vzero;
      int krow = (c * 16 + lr) * 72;
#pragma unroll
      for (int f = 0; f < 2; ++f) {
        Frag16 bk;     // B 32x16: lane<16 K(dh) {0-7,16-23}, else {8-15,24-31} of this dh slice
        int k0 = f * 32 + half * 8;
        bk.h[0] = *reinterpret_cast<const v8bf*>(&Ks[buf][krow + k0]);
        bk.h[1] = *reinterpret_cast<const v8bf*>(&Ks[buf][krow + k0 + 16]);
        sc[c] = __builtin_amdgcn_wmma_f32_16x16x32_bf16(
            false, aq[f].v, false, bk.v, (short)0, sc[c], false, false);
      }
    }
    // online softmax (row stats per lane-half: vgpr j <-> row j / j+8)
    float mnew[8], scale[8];
#pragma unroll
    for (int j = 0; j < 8; ++j) {
      float mx = fmaxf(sc[0][j], sc[1][j]);
#pragma unroll
      for (int sh = 1; sh < 16; sh <<= 1) mx = fmaxf(mx, __shfl_xor(mx, sh));
      mnew[j]  = fmaxf(mrow[j], mx);
      scale[j] = __expf(mrow[j] - mnew[j]);
      mrow[j]  = mnew[j];
    }
#pragma unroll
    for (int j = 0; j < 8; ++j) {
      sc[0][j] = __expf(sc[0][j] - mnew[j]);
      sc[1][j] = __expf(sc[1][j] - mnew[j]);
      float rs = sc[0][j] + sc[1][j];
#pragma unroll
      for (int sh = 1; sh < 16; sh <<= 1) rs += __shfl_xor(rs, sh);
      lrow[j] = lrow[j] * scale[j] + rs;
#pragma unroll
      for (int d = 0; d < 4; ++d) o[d][j] *= scale[j];
    }
    // transpose P: C-layout -> A-layout through per-wave LDS tile
#pragma unroll
    for (int c = 0; c < 2; ++c)
#pragma unroll
      for (int j = 0; j < 8; ++j)
        pb[(j + half * 8) * 40 + c * 16 + lr] = f2bf(sc[c][j]);
    Frag16 pf;
    {
      int k0 = half * 8;
      pf.h[0] = *reinterpret_cast<const v8bf*>(&pb[lr * 40 + k0]);
      pf.h[1] = *reinterpret_cast<const v8bf*>(&pb[lr * 40 + k0 + 16]);
    }
    // O += P x V from LDS-staged V^T
#pragma unroll
    for (int d = 0; d < 4; ++d) {
      Frag16 bv;
      int vrow = (d * 16 + lr) * 40 + half * 8;
      bv.h[0] = *reinterpret_cast<const v8bf*>(&Vs[buf][vrow]);
      bv.h[1] = *reinterpret_cast<const v8bf*>(&Vs[buf][vrow + 16]);
      o[d] = __builtin_amdgcn_wmma_f32_16x16x32_bf16(
          false, pf.v, false, bv.v, (short)0, o[d], false, false);
    }
  };

  const int NKB = S_ / 32;   // 32 chunks (even)
  stage(0, 0);
  wait_async0();
  __syncthreads();

  for (int kc = 0; kc < NKB - 2; kc += 2) {
    stage((kc + 1) * 32, 1);
    chunk(0);
    wait_async0(); __syncthreads();
    stage((kc + 2) * 32, 0);
    chunk(1);
    wait_async0(); __syncthreads();
  }
  stage((NKB - 1) * 32, 1);
  chunk(0);
  wait_async0(); __syncthreads();
  chunk(1);

  // normalize + store fp32 output
  float inv[8];
#pragma unroll
  for (int j = 0; j < 8; ++j) inv[j] = 1.0f / lrow[j];
#pragma unroll
  for (int d = 0; d < 4; ++d)
#pragma unroll
    for (int j = 0; j < 8; ++j) {
      int m = j + half * 8;
      out[((size_t)(b * S_ + q0 + m)) * OD_ + h * DH_ + d * 16 + lr] =
          o[d][j] * inv[j];
    }
}

// ---------------- host launcher ----------------
extern "C" void kernel_launch(void* const* d_in, const int* in_sizes, int n_in,
                              void* d_out, int out_size, void* d_ws, size_t ws_size,
                              hipStream_t stream) {
  (void)in_sizes; (void)n_in; (void)out_size; (void)ws_size;
  const float* x  = (const float*)d_in[0];
  const float* qw = (const float*)d_in[1];
  const float* kw = (const float*)d_in[2];
  const float* vw = (const float*)d_in[3];
  float* out = (float*)d_out;

  char* ws = (char*)d_ws;
  // workspace layout (bytes), all offsets 256-aligned; total ~64.5 MB
  bf16* xb = (bf16*)(ws + 0);                                   // 8192*768*2
  bf16* wt = (bf16*)(ws + (size_t)12582912);                    // 3072*768*2
  bf16* Qb = (bf16*)(ws + (size_t)17301504);                    // 8192*1024*2
  bf16* Kb = (bf16*)(ws + (size_t)34078720);                    // 8192*1024*2
  bf16* Vt = (bf16*)(ws + (size_t)50855936);                    // 8192*1024*2

  k_conv_x<<<(M_ * D_) / 4 / 256, 256, 0, stream>>>(x, xb);
  k_conv_w<<<(3 * D_ * OD_) / 256, 256, 0, stream>>>(qw, kw, vw, wt);
  k_qkv_gemm<<<dim3(M_ / 128, N3_ / 128), 256, 0, stream>>>(xb, wt, Qb, Kb, Vt);
  k_attn<<<dim3(B_ * H_, S_ / 128), 256, 0, stream>>>(Qb, Kb, Vt, out);
}
